// LayerNormGRU_28948079575316
// MI455X (gfx1250) — compile-verified, matrix operand order checked by reading, not run on hardware
//
#include <hip/hip_runtime.h>
#include <hip/hip_bf16.h>

typedef __attribute__((ext_vector_type(16))) _Float16 v16h;
typedef __attribute__((ext_vector_type(8)))  float    v8f;
typedef __attribute__((ext_vector_type(4)))  unsigned int u32x4;
typedef __attribute__((ext_vector_type(8)))  int      i32x8;
typedef __attribute__((ext_vector_type(4)))  int      i32x4;

#define T_  512
#define B_  64
#define D_  512
#define H_  512
#define L_  2
#define G3H (3*H_)
#define EPS 1e-5f

// --------------------------------------------------------------------------
// Fragment position for a 16-bit WMMA A/B operand element (ISA 7.12.2).
//   r<4 : k = 2r + p + 8*half   ;  r>=4: k = 16 + 2(r-4) + p + 8*half
//   lane = half*16 + row16 ; slot = 2r + p
// --------------------------------------------------------------------------
__device__ __host__ __forceinline__ void frag_pos(int rc16, int kk, int& lane, int& slot) {
    int base = (kk & 16) ? 4 : 0;
    int k2   = kk & 15;
    int half = k2 >> 3;
    int rem  = k2 & 7;
    int r    = (rem >> 1) + base;
    int p    = rem & 1;
    lane = half * 16 + rc16;
    slot = r * 2 + p;
}

// Pack a row-major fp32 matrix (M x K) into f16 fragment order:
// out[((tm*KT + kt)*32 + lane)*16 + slot].
__global__ void pack_frag_kernel(const float* __restrict__ src,
                                 _Float16* __restrict__ dst,
                                 int M, int K) {
    long long idx = (long long)blockIdx.x * blockDim.x + threadIdx.x;
    long long total = (long long)M * K;
    if (idx >= total) return;
    int m = (int)(idx / K);
    int k = (int)(idx % K);
    int lane, slot;
    frag_pos(m & 15, k & 31, lane, slot);
    int tm = m >> 4, kt = k >> 5, KT = K >> 5;
    dst[(((long long)tm * KT + kt) * 32 + lane) * 16 + slot] = (_Float16)src[idx];
}

// --------------------------------------------------------------------------
// GEMM: C(M x N) = Afrag * Bfrag + bias, fp32 accumulate.
// One wave computes a 16x64 strip: 4 WMMA accumulators share each A fragment
// (4x less A traffic). Operands pre-packed so every load is 32B/lane aligned.
// --------------------------------------------------------------------------
__global__ void __launch_bounds__(32)
gemm_frag_bias_kernel(const _Float16* __restrict__ Af,
                      const _Float16* __restrict__ Bf,
                      const float* __restrict__ bias,
                      float* __restrict__ C,
                      int N, int KT) {
    int tm  = blockIdx.x;
    int tn0 = blockIdx.y * 4;
    int lane = threadIdx.x;
    const long long bstride = (long long)KT * 32 * 16;    // halves per n-tile
    const _Float16* ap = Af + (((long long)tm  * KT) * 32 + lane) * 16;
    const _Float16* bp = Bf + (((long long)tn0 * KT) * 32 + lane) * 16;
    v8f acc[4] = {};
    for (int kt = 0; kt < KT; ++kt) {
        v16h a  = *(const v16h*)ap;                     ap += 32 * 16;
        v16h b0 = *(const v16h*)(bp);
        v16h b1 = *(const v16h*)(bp + bstride);
        v16h b2 = *(const v16h*)(bp + 2 * bstride);
        v16h b3 = *(const v16h*)(bp + 3 * bstride);
        __builtin_prefetch(bp + 2 * 32 * 16, 0, 1);     // global_prefetch_b8
        bp += 32 * 16;
        acc[0] = __builtin_amdgcn_wmma_f32_16x16x32_f16(false, a, false, b0, (short)0, acc[0], false, false);
        acc[1] = __builtin_amdgcn_wmma_f32_16x16x32_f16(false, a, false, b1, (short)0, acc[1], false, false);
        acc[2] = __builtin_amdgcn_wmma_f32_16x16x32_f16(false, a, false, b2, (short)0, acc[2], false, false);
        acc[3] = __builtin_amdgcn_wmma_f32_16x16x32_f16(false, a, false, b3, (short)0, acc[3], false, false);
    }
    int m0 = tm * 16 + ((lane >> 4) << 3);   // C layout: vgpr v -> M = v (+8 for lanes 16..31)
    #pragma unroll
    for (int u = 0; u < 4; ++u) {
        union { v8f v; float f[8]; } r; r.v = acc[u];
        int n = (tn0 + u) * 16 + (lane & 15);
        float bn = bias[n];
        for (int v = 0; v < 8; ++v)
            C[(long long)(m0 + v) * N + n] = r.f[v] + bn;
    }
}

// --------------------------------------------------------------------------
// Persistent recurrence: one workgroup (32 waves) runs all T steps of one
// layer. h lives in LDS (fp32 copy + f16 copy in WMMA A-fragment order,
// 192 KB total, fits the 320 KB WGP LDS). Initial h is DMA'd in via the
// Tensor Data Mover when available.
// --------------------------------------------------------------------------
__global__ void __launch_bounds__(1024)
recur_kernel(const float* __restrict__ GI,       // (T*B, 3H), bih already added
             const _Float16* __restrict__ WhF,   // packed Wh fragments
             const float* __restrict__ bhh,      // (3H)
             const float* __restrict__ lng,      // (3,H)
             const float* __restrict__ lnb,      // (3,H)
             const float* __restrict__ h0,       // (B,H)
             float* __restrict__ Yout,           // (T*B, H)
             float* __restrict__ hlast,          // (B,H)
             float* __restrict__ GH,             // scratch (B,3H)
             float* __restrict__ Rg,             // scratch (B,H)
             float* __restrict__ Ig) {           // scratch (B,H)
    extern __shared__ char smem[];
    float*    hf32  = (float*)smem;                        // B*H fp32  = 128 KB (LDS offset 0)
    _Float16* hfrag = (_Float16*)(smem + B_ * H_ * 4);     // B*H f16   =  64 KB
    const int tid  = threadIdx.x;
    const int lane = tid & 31;
    const int wave = tid >> 5;
    const int KT   = H_ >> 5;      // 16 k-tiles
    const long long bstride = (long long)KT * 32 * 16;

    // ---- Load h0 into LDS (fp32 image) -----------------------------------
#if __has_builtin(__builtin_amdgcn_tensor_load_to_lds)
    if (tid < 32) {
        // Tensor DMA descriptor (ISA 8.3/8.4): 1D tile, 32768 x 4B elements,
        // global h0 -> LDS offset 0 (hf32).
        unsigned long long ga = (unsigned long long)(const void*)h0;
        u32x4 g0 = {};
        g0[0] = 1u;                                            // count=1 (valid user D#)
        g0[1] = 0u;                                            // lds_addr = 0
        g0[2] = (unsigned)(ga & 0xFFFFFFFFu);                  // global_addr[31:0]
        g0[3] = (unsigned)((ga >> 32) & 0x01FFFFFFu) | (2u << 30); // addr[56:32] | type=2
        const unsigned td0 = (unsigned)(B_ * H_);              // 32768 elements
        i32x8 g1 = {};
        g1[0] = (int)(2u << 16);                               // data_size = 4 bytes
        g1[1] = (int)((td0 & 0xFFFFu) << 16);                  // tensor_dim0[15:0]
        g1[2] = (int)((td0 >> 16) & 0xFFFFu) | (1 << 16);      // tensor_dim0[31:16], tensor_dim1=1
        g1[3] = (int)((td0 & 0xFFFFu) << 16);                  // tile_dim0 = 32768
        g1[4] = 1;                                             // tile_dim1 = 1
        g1[5] = (int)td0;                                      // tensor_dim0_stride = 32768
        i32x4 g2 = {}, g3 = {};
#if defined(__clang_major__) && (__clang_major__ >= 23)
        i32x8 g4 = {};
        __builtin_amdgcn_tensor_load_to_lds(g0, g1, g2, g3, g4, 0);
#else
        __builtin_amdgcn_tensor_load_to_lds(g0, g1, g2, g3, 0);
#endif
        __builtin_amdgcn_s_wait_tensorcnt(0);
    }
#else
    for (int i = tid; i < B_ * H_; i += 1024) hf32[i] = h0[i];
#endif
    __syncthreads();

    // Build the f16 fragment image of h from the fp32 LDS copy.
    for (int i = tid; i < B_ * H_; i += 1024) {
        float v = hf32[i];
        int b = i >> 9, j = i & (H_ - 1);
        int ln2, sl; frag_pos(b & 15, j & 31, ln2, sl);
        hfrag[((((b >> 4) * KT) + (j >> 5)) * 32 + ln2) * 16 + sl] = (_Float16)v;
    }
    __syncthreads();

    // Tile assignment: each wave owns one M-tile row and 12 N-tiles,
    // processed as 3 groups of 4 so each A fragment is reused 4x per read.
    const int tmw    = wave >> 3;        // 0..3
    const int tnbase = (wave & 7) * 12;  // 0,12,...,84

    for (int t = 0; t < T_; ++t) {
        const float* GIt = GI + (long long)t * B_ * G3H;

        // ---- Phase A: GH = h @ Wh.T + bhh
        for (int g = 0; g < 3; ++g) {
            int tn0 = tnbase + g * 4;
            const _Float16* ap = hfrag + (((long long)tmw * KT) * 32 + lane) * 16;
            const _Float16* bp = WhF   + (((long long)tn0 * KT) * 32 + lane) * 16;
            v8f acc[4] = {};
            for (int kt = 0; kt < KT; ++kt) {
                v16h a  = *(const v16h*)ap;                 ap += 32 * 16;
                v16h b0 = *(const v16h*)(bp);
                v16h b1 = *(const v16h*)(bp + bstride);
                v16h b2 = *(const v16h*)(bp + 2 * bstride);
                v16h b3 = *(const v16h*)(bp + 3 * bstride);
                __builtin_prefetch(bp + 2 * 32 * 16, 0, 1);
                bp += 32 * 16;
                acc[0] = __builtin_amdgcn_wmma_f32_16x16x32_f16(false, a, false, b0, (short)0, acc[0], false, false);
                acc[1] = __builtin_amdgcn_wmma_f32_16x16x32_f16(false, a, false, b1, (short)0, acc[1], false, false);
                acc[2] = __builtin_amdgcn_wmma_f32_16x16x32_f16(false, a, false, b2, (short)0, acc[2], false, false);
                acc[3] = __builtin_amdgcn_wmma_f32_16x16x32_f16(false, a, false, b3, (short)0, acc[3], false, false);
            }
            int b0r = tmw * 16 + ((lane >> 4) << 3);
            #pragma unroll
            for (int u = 0; u < 4; ++u) {
                union { v8f v; float f[8]; } r; r.v = acc[u];
                int n = (tn0 + u) * 16 + (lane & 15);
                float bn = bhh[n];
                for (int v = 0; v < 8; ++v)
                    GH[(b0r + v) * G3H + n] = r.f[v] + bn;
            }
        }
        __syncthreads();

        // ---- Phase B: reset & input gates: sigmoid(LN(i_g + h_g))
        for (int q = 0; q < 4; ++q) {
            int row = wave * 4 + q;          // 0..127
            int b = row >> 1, g = row & 1;
            const float* gi = GIt + (long long)b * G3H + g * H_;
            const float* gh = GH + b * G3H + g * H_;
            float vv[16], s = 0.f, ss = 0.f;
            for (int jj = 0; jj < 16; ++jj) {
                int j = jj * 32 + lane;
                float v = gi[j] + gh[j];
                vv[jj] = v; s += v; ss += v * v;
            }
            for (int m = 16; m; m >>= 1) {
                s  += __shfl_xor(s,  m, 32);
                ss += __shfl_xor(ss, m, 32);
            }
            float mu  = s * (1.f / H_);
            float var = ss * (1.f / H_) - mu * mu;
            float rs  = rsqrtf(var + EPS);
            float* outp = (g == 0) ? Rg : Ig;
            for (int jj = 0; jj < 16; ++jj) {
                int j = jj * 32 + lane;
                float y = (vv[jj] - mu) * rs * lng[g * H_ + j] + lnb[g * H_ + j];
                outp[b * H_ + j] = 1.f / (1.f + __expf(-y));
            }
        }
        __syncthreads();

        // ---- Phase C: new gate + state update
        for (int q = 0; q < 2; ++q) {
            int b = wave * 2 + q;
            const float* gi = GIt + (long long)b * G3H + 2 * H_;
            const float* gh = GH + b * G3H + 2 * H_;
            float vv[16], s = 0.f, ss = 0.f;
            for (int jj = 0; jj < 16; ++jj) {
                int j = jj * 32 + lane;
                float v = gi[j] + Rg[b * H_ + j] * gh[j];
                vv[jj] = v; s += v; ss += v * v;
            }
            for (int m = 16; m; m >>= 1) {
                s  += __shfl_xor(s,  m, 32);
                ss += __shfl_xor(ss, m, 32);
            }
            float mu  = s * (1.f / H_);
            float var = ss * (1.f / H_) - mu * mu;
            float rs  = rsqrtf(var + EPS);
            for (int jj = 0; jj < 16; ++jj) {
                int j = jj * 32 + lane;
                float y  = (vv[jj] - mu) * rs * lng[2 * H_ + j] + lnb[2 * H_ + j];
                float ng = tanhf(y);
                float hold = hf32[b * H_ + j];
                float ig   = Ig[b * H_ + j];
                float hy   = ng + ig * (hold - ng);
                hf32[b * H_ + j] = hy;
                int ln2, sl; frag_pos(b & 15, j & 31, ln2, sl);
                hfrag[((((b >> 4) * KT) + (j >> 5)) * 32 + ln2) * 16 + sl] = (_Float16)hy;
                Yout[((long long)t * B_ + b) * H_ + j] = hy;
                if (t == T_ - 1) hlast[b * H_ + j] = hy;
            }
        }
        __syncthreads();
    }
}

// --------------------------------------------------------------------------
extern "C" void kernel_launch(void* const* d_in, const int* in_sizes, int n_in,
                              void* d_out, int out_size, void* d_ws, size_t ws_size,
                              hipStream_t stream) {
    const float* X   = (const float*)d_in[0];   // (T,B,D)
    const float* Hid = (const float*)d_in[1];   // (L,B,H)
    const float* Wih = (const float*)d_in[2];   // (L,3H,D)
    const float* bih = (const float*)d_in[3];   // (L,3H)
    const float* Whh = (const float*)d_in[4];   // (L,3H,H)
    const float* bhh = (const float*)d_in[5];   // (L,3H)
    const float* lng = (const float*)d_in[6];   // (L,3,H)
    const float* lnb = (const float*)d_in[7];   // (L,3,H)
    float* out = (float*)d_out;                  // x (T*B*H) ++ hiddens (L*B*H)

    // Workspace carve
    char* w = (char*)d_ws;
    float* GI = (float*)w;  w += (size_t)T_ * B_ * G3H * 4;     // 192 MB
    float* GH = (float*)w;  w += (size_t)B_ * G3H * 4;
    float* Rg = (float*)w;  w += (size_t)B_ * H_ * 4;
    float* Ig = (float*)w;  w += (size_t)B_ * H_ * 4;
    float* Y0 = (float*)w;  w += (size_t)T_ * B_ * H_ * 4;      // 64 MB
    _Float16* Ap = (_Float16*)w;  w += (size_t)T_ * B_ * D_ * 2; // 32 MB
    _Float16* WiP[L_], *WhP[L_];
    for (int l = 0; l < L_; ++l) { WiP[l] = (_Float16*)w; w += (size_t)G3H * D_ * 2; }
    for (int l = 0; l < L_; ++l) { WhP[l] = (_Float16*)w; w += (size_t)G3H * H_ * 2; }

    hipFuncSetAttribute((const void*)recur_kernel,
                        hipFuncAttributeMaxDynamicSharedMemorySize,
                        B_ * H_ * 4 + B_ * H_ * 2);

    const long long wElems = (long long)G3H * D_;
    const int PB = 256;
    for (int l = 0; l < L_; ++l) {
        pack_frag_kernel<<<dim3((unsigned)((wElems + PB - 1) / PB)), PB, 0, stream>>>(
            Wih + (long long)l * G3H * D_, WiP[l], G3H, D_);
        pack_frag_kernel<<<dim3((unsigned)((wElems + PB - 1) / PB)), PB, 0, stream>>>(
            Whh + (long long)l * G3H * H_, WhP[l], G3H, H_);
    }

    const long long aElems = (long long)T_ * B_ * D_;
    const size_t shmem = B_ * H_ * 4 + B_ * H_ * 2;   // 192 KB dynamic LDS

    for (int l = 0; l < L_; ++l) {
        const float* Xl = (l == 0) ? X : Y0;
        pack_frag_kernel<<<dim3((unsigned)((aElems + PB - 1) / PB)), PB, 0, stream>>>(
            Xl, Ap, T_ * B_, D_);
        gemm_frag_bias_kernel<<<dim3((T_ * B_) / 16, G3H / 64), 32, 0, stream>>>(
            Ap, WiP[l], bih + l * G3H, GI, G3H, D_ / 32);
        float* Yl    = (l == L_ - 1) ? out : Y0;
        float* hlast = out + (size_t)T_ * B_ * H_ + (size_t)l * B_ * H_;
        recur_kernel<<<1, 1024, shmem, stream>>>(
            GI, WhP[l], bhh + l * G3H,
            lng + (size_t)l * 3 * H_, lnb + (size_t)l * 3 * H_,
            Hid + (size_t)l * B_ * H_,
            Yl, hlast, GH, Rg, Ig);
    }
}
